// HGSABlock_24352464569609
// MI455X (gfx1250) — compile-verified, matrix operand order checked by reading, not run on hardware
//
#include <hip/hip_runtime.h>

// ---------------------------------------------------------------------------
// HGSA-style network forward pass for gfx1250 (MI455X).
// All 1x1 convs (the dominant FLOPs) run through v_wmma_f32_16x16x32_f16.
// Everything compile-time specialized: no runtime div/mod, no branchy epilogue.
// ---------------------------------------------------------------------------

#define NTOK   147456        // 4 * 192 * 192 tokens
#define HWPIX  36864         // 192 * 192
#define BATCH  4
#define IMGW   192
#define IMGH   192

typedef __attribute__((ext_vector_type(16))) _Float16 v16h;
typedef __attribute__((ext_vector_type(8)))  float    v8f;

// act codes: 0=none 1=gelu(exact erf) 2=sigmoid 3=silu  (compile-time)
template<int ACT>
__device__ __forceinline__ float act_f(float x) {
    if (ACT == 1) return 0.5f * x * (1.0f + erff(x * 0.7071067811865475f));
    if (ACT == 2) return __fdividef(1.0f, 1.0f + __expf(-x));
    if (ACT == 3) return x * __fdividef(1.0f, 1.0f + __expf(-x));
    return x;
}

// ---------------------------------------------------------------------------
// Weight pre-pad: W[N][K] -> Wp[Np][Kp], zero-filled (Np=ceil16, Kp=ceil32)
// ---------------------------------------------------------------------------
__global__ void k_padw(const float* __restrict__ W, float* __restrict__ Wp,
                       int N, int K, int Kp, int total)
{
    int i = blockIdx.x * 256 + threadIdx.x;
    if (i >= total) return;
    int col = i % Kp;
    int row = i / Kp;
    Wp[i] = (row < N && col < K) ? W[row * K + col] : 0.f;
}

// ---------------------------------------------------------------------------
// WMMA GEMM: Y[rows, N] = act(A[rows, lda(>=KP)] * Wp[Np, KP]^T + bias)
// A zero-padded beyond K; Wp zero-padded. Unconditional loads; A fragments
// converted once into registers and reused across all N column tiles.
// One wave -> one 16x16 tile row; 8 waves/block -> 128 rows/block.
// ---------------------------------------------------------------------------
template<int KP, int ACT, bool HASBIAS>
__global__ __launch_bounds__(256)
void k_gemm_t(const float* __restrict__ A, int lda,
              const float* __restrict__ Wp,
              const float* __restrict__ bias, float* __restrict__ Y, int N)
{
    const int lane    = threadIdx.x & 31;
    const int wave    = threadIdx.x >> 5;
    const int rowTile = blockIdx.x * 8 + wave;      // 16 rows per tile
    const int half    = lane >> 4;                  // K sub-range selector
    const size_t arow = (size_t)(rowTile * 16 + (lane & 15)) * lda;
    const int colTiles = (N + 15) >> 4;

    // stage converted A fragments in registers (KP/32 x 8 VGPRs)
    v16h areg[KP / 32];
    #pragma unroll
    for (int ks = 0; ks < KP / 32; ++ks) {
        #pragma unroll
        for (int v = 0; v < 8; ++v) {
            // 16-bit A layout: VGPR v holds K = (v<4?0:16) + half*8 + (v&3)*2, +1
            int kk = ks * 32 + ((v & 4) << 2) + half * 8 + ((v & 3) << 1);
            float2 av = *(const float2*)(A + arow + kk);
            areg[ks][2 * v]     = (_Float16)av.x;
            areg[ks][2 * v + 1] = (_Float16)av.y;
        }
    }

    for (int ct = 0; ct < colTiles; ++ct) {
        v8f acc = {0.f, 0.f, 0.f, 0.f, 0.f, 0.f, 0.f, 0.f};
        const size_t brow = (size_t)(ct * 16 + (lane & 15)) * KP;
        #pragma unroll
        for (int ks = 0; ks < KP / 32; ++ks) {
            v16h b;
            #pragma unroll
            for (int v = 0; v < 8; ++v) {
                int kk = ks * 32 + ((v & 4) << 2) + half * 8 + ((v & 3) << 1);
                float2 bv = *(const float2*)(Wp + brow + kk);
                b[2 * v]     = (_Float16)bv.x;
                b[2 * v + 1] = (_Float16)bv.y;
            }
            acc = __builtin_amdgcn_wmma_f32_16x16x32_f16(
                      false, areg[ks], false, b, (short)0, acc, false, false);
        }
        #pragma unroll
        for (int v = 0; v < 8; ++v) {
            int mr = rowTile * 16 + v + 8 * half;   // C/D layout: M = vgpr + 8*half
            int nc = ct * 16 + (lane & 15);         // N = lane%16
            if (nc < N) {
                float val = acc[v];
                if (HASBIAS) val += bias[nc];
                Y[(size_t)mr * N + nc] = act_f<ACT>(val);
            }
        }
    }
}

// ---------------------------------------------------------------------------
// Depthwise conv (SAME), channels-last [B,H,W,C], weight [C][KS*KS], fused act
// ---------------------------------------------------------------------------
template<int C, int KS, int ACT, bool HASBIAS>
__global__ void k_dwconv_t(const float* __restrict__ x, const float* __restrict__ w,
                           const float* __restrict__ bias, float* __restrict__ y)
{
    size_t i = (size_t)blockIdx.x * 256 + threadIdx.x;
    if (i >= (size_t)NTOK * C) return;
    int c = (int)(i % C);
    size_t pix = i / C;
    int wx = (int)(pix % IMGW);
    int hy = (int)((pix / IMGW) % IMGH);
    int b  = (int)(pix / HWPIX);
    const int r = KS >> 1;
    float s = HASBIAS ? bias[c] : 0.f;
    #pragma unroll
    for (int dy = -r; dy <= r; ++dy) {
        int yy = hy + dy; if (yy < 0 || yy >= IMGH) continue;
        #pragma unroll
        for (int dx = -r; dx <= r; ++dx) {
            int xx = wx + dx; if (xx < 0 || xx >= IMGW) continue;
            s += x[(((size_t)b * IMGH + yy) * IMGW + xx) * C + c]
               * w[c * KS * KS + (dy + r) * KS + (dx + r)];
        }
    }
    y[i] = act_f<ACT>(s);
}

// LayerNorm over C=32: one wave per token, lane==channel, shfl reductions
__global__ __launch_bounds__(256)
void k_layernorm(const float* __restrict__ x, const float* __restrict__ g,
                 const float* __restrict__ bb, float* __restrict__ y)
{
    int lane = threadIdx.x & 31;
    int wv   = threadIdx.x >> 5;
    size_t tok = (size_t)blockIdx.x * 8 + wv;
    float v = x[tok * 32 + lane];
    float s = v;
    for (int o = 16; o > 0; o >>= 1) s += __shfl_xor(s, o, 32);
    float mean = s * (1.f / 32.f);
    float d = v - mean;
    float vs = d * d;
    for (int o = 16; o > 0; o >>= 1) vs += __shfl_xor(vs, o, 32);
    float var = vs * (1.f / 32.f);
    y[tok * 32 + lane] = d * rsqrtf(var + 1e-5f) * g[lane] + bb[lane];
}

// per-(b,c) L2 norm over the n (token) axis; grid = B*32
__global__ __launch_bounds__(256)
void k_colnorm(const float* __restrict__ x, float* __restrict__ norms)
{
    int bc = blockIdx.x;
    int b = bc >> 5, c = bc & 31;
    __shared__ float red[256];
    float s = 0.f;
    const float* xb = x + (size_t)b * HWPIX * 32 + c;
    for (int n = threadIdx.x; n < HWPIX; n += 256) {
        float v = xb[(size_t)n * 32];
        s += v * v;
    }
    red[threadIdx.x] = s; __syncthreads();
    for (int st = 128; st > 0; st >>= 1) {
        if ((int)threadIdx.x < st) red[threadIdx.x] += red[threadIdx.x + st];
        __syncthreads();
    }
    if (threadIdx.x == 0) norms[bc] = fmaxf(sqrtf(red[0]), 1e-12f);
}

// Gram matrix G[d,e] = sum_n khat[n,cd]*qhat[n,ce], rescale, softmax over e.
// grid = B*HEADS, block = 256. dh = 32/HEADS (16 or 8).
template<int HEADS>
__global__ __launch_bounds__(256)
void k_gram(const float* __restrict__ q, const float* __restrict__ k,
            const float* __restrict__ qn, const float* __restrict__ kn,
            const float* __restrict__ rescale, float* __restrict__ attn)
{
    const int C = 32;
    const int dh = C / HEADS;
    const int pairs = dh * dh;
    const int chunks = 256 / pairs;
    const int bh = blockIdx.x;
    const int b = bh / HEADS, h = bh % HEADS;
    __shared__ float red[256];
    __shared__ float gmat[256];
    int t = threadIdx.x;
    int p = t % pairs, cchunk = t / pairs;
    int dd = p / dh, ee = p % dh;
    int cd = h * dh + dd, ce = h * dh + ee;
    const float* kb = k + (size_t)b * HWPIX * C;
    const float* qb = q + (size_t)b * HWPIX * C;
    float s = 0.f;
    for (int n = cchunk; n < HWPIX; n += chunks)
        s += kb[(size_t)n * C + cd] * qb[(size_t)n * C + ce];
    red[t] = s; __syncthreads();
    if (cchunk == 0) {
        #pragma unroll
        for (int cc = 1; cc < chunks; ++cc) s += red[p + cc * pairs];
        s = s / (kn[b * C + cd] * qn[b * C + ce]) * rescale[h];
        gmat[p] = s;
    }
    __syncthreads();
    if (t < dh) {
        float mx = -1e30f;
        #pragma unroll
        for (int e2 = 0; e2 < dh; ++e2) mx = fmaxf(mx, gmat[t * dh + e2]);
        float sum = 0.f;
        #pragma unroll
        for (int e2 = 0; e2 < dh; ++e2) sum += __expf(gmat[t * dh + e2] - mx);
        float inv = __fdividef(1.f, sum);
        #pragma unroll
        for (int e2 = 0; e2 < dh; ++e2)
            attn[((size_t)bh * dh + t) * dh + e2] = __expf(gmat[t * dh + e2] - mx) * inv;
    }
}

// out[tok, h*dh+d] = sum_e attn[b,h,d,e] * v[tok, h*dh+e]
template<int HEADS>
__global__ void k_apply_attn(const float* __restrict__ attn, const float* __restrict__ v,
                             float* __restrict__ out)
{
    const int C = 32;
    const int dh = C / HEADS;
    size_t i = (size_t)blockIdx.x * 256 + threadIdx.x;
    if (i >= (size_t)NTOK * C) return;
    int c = (int)(i % C);
    size_t tok = i / C;
    int b = (int)(tok / HWPIX);
    int h = c / dh, dd = c % dh;
    const float* am = attn + ((size_t)(b * HEADS + h) * dh + dd) * dh;
    const float* vr = v + tok * C + h * dh;
    float s = 0.f;
    #pragma unroll
    for (int e = 0; e < dh; ++e) s += am[e] * vr[e];
    out[i] = s;
}

// ------------------------- elementwise helpers -----------------------------
template<int C>
__global__ void k_tocl(const float* __restrict__ src, float* __restrict__ dst) {
    size_t i = (size_t)blockIdx.x * 256 + threadIdx.x;
    if (i >= (size_t)NTOK * C) return;
    int c = (int)(i % C);
    size_t t = i / C;
    int n = (int)(t % HWPIX);
    int b = (int)(t / HWPIX);
    dst[i] = src[((size_t)b * C + c) * HWPIX + n];
}
// concat with padded output stride LDC (zero fill for c in [ca+cb, LDC))
template<int LDC>
__global__ void k_concat2(const float* __restrict__ a, int ca,
                          const float* __restrict__ b2, int cb,
                          float* __restrict__ dst) {
    size_t i = (size_t)blockIdx.x * 256 + threadIdx.x;
    if (i >= (size_t)NTOK * LDC) return;
    int c = (int)(i % LDC);
    size_t t = i / LDC;
    float v = 0.f;
    if (c < ca)           v = a[t * ca + c];
    else if (c < ca + cb) v = b2[t * cb + (c - ca)];
    dst[i] = v;
}
template<int LDC>
__global__ void k_concat3(const float* __restrict__ a, int ca,
                          const float* __restrict__ b2, int cb,
                          const float* __restrict__ c3, int cc,
                          float* __restrict__ dst) {
    size_t i = (size_t)blockIdx.x * 256 + threadIdx.x;
    if (i >= (size_t)NTOK * LDC) return;
    int c = (int)(i % LDC);
    size_t t = i / LDC;
    float v = 0.f;
    if (c < ca)                v = a[t * ca + c];
    else if (c < ca + cb)      v = b2[t * cb + (c - ca)];
    else if (c < ca + cb + cc) v = c3[t * cc + (c - ca - cb)];
    dst[i] = v;
}
__global__ void k_copy(float* __restrict__ d, const float* __restrict__ s, size_t n) {
    size_t i = (size_t)blockIdx.x * 256 + threadIdx.x;
    if (i < n) d[i] = s[i];
}
__global__ void k_zero(float* __restrict__ d, size_t n) {
    size_t i = (size_t)blockIdx.x * 256 + threadIdx.x;
    if (i < n) d[i] = 0.f;
}
__global__ void k_add1(float* __restrict__ x, const float* __restrict__ a, size_t n) {
    size_t i = (size_t)blockIdx.x * 256 + threadIdx.x;
    if (i < n) x[i] += a[i];
}
__global__ void k_add2(float* __restrict__ x, const float* __restrict__ a,
                       const float* __restrict__ b, size_t n) {
    size_t i = (size_t)blockIdx.x * 256 + threadIdx.x;
    if (i < n) x[i] += a[i] + b[i];
}
// v *= (m1*am + m1)   (MaskGuidedMechanism modulation)
__global__ void k_vmul_ma(float* __restrict__ v, const float* __restrict__ m1,
                          const float* __restrict__ am, size_t n) {
    size_t i = (size_t)blockIdx.x * 256 + threadIdx.x;
    if (i < n) v[i] *= (m1[i] * am[i] + m1[i]);
}

// basis attention: softmax over 16 of q*temp, project through basis_v[16,7]
__global__ void k_basis(const float* __restrict__ q16, const float* __restrict__ basis_v,
                        const float* __restrict__ temp, float* __restrict__ xi) {
    size_t t = (size_t)blockIdx.x * 256 + threadIdx.x;
    if (t >= NTOK) return;
    float tv = temp[0];
    float vals[16];
    float mx = -1e30f;
    #pragma unroll
    for (int i = 0; i < 16; ++i) { vals[i] = q16[t * 16 + i] * tv; mx = fmaxf(mx, vals[i]); }
    float sum = 0.f;
    #pragma unroll
    for (int i = 0; i < 16; ++i) { vals[i] = __expf(vals[i] - mx); sum += vals[i]; }
    float inv = __fdividef(1.f, sum);
    #pragma unroll
    for (int j = 0; j < 7; ++j) {
        float s = 0.f;
        #pragma unroll
        for (int i = 0; i < 16; ++i) s += basis_v[i * 7 + j] * vals[i];
        xi[t * 7 + j] = s * inv;
    }
}

// data[tok,70] -> gates (softmax over M=5) and taus; channel = m*7+q
__global__ void k_gates(const float* __restrict__ data, float* __restrict__ gates,
                        float* __restrict__ taus) {
    size_t i = (size_t)blockIdx.x * 256 + threadIdx.x;
    if (i >= (size_t)NTOK * 7) return;
    int qc = (int)(i % 7);
    size_t t = i / 7;
    float vals[5];
    float mx = -1e30f;
    #pragma unroll
    for (int m = 0; m < 5; ++m) { vals[m] = data[t * 70 + m * 7 + qc]; mx = fmaxf(mx, vals[m]); }
    float sum = 0.f;
    #pragma unroll
    for (int m = 0; m < 5; ++m) { vals[m] = __expf(vals[m] - mx); sum += vals[m]; }
    float inv = __fdividef(1.f, sum);
    #pragma unroll
    for (int m = 0; m < 5; ++m) {
        gates[t * 35 + m * 7 + qc] = vals[m] * inv;
        taus[t * 35 + m * 7 + qc]  = data[t * 70 + 35 + m * 7 + qc];
    }
}

// psi += gates_i * w * exp(-0.5*((xi-mu)/sigma)^2) for expert ei
__global__ void k_expert_acc(const float* __restrict__ sc, const float* __restrict__ mb,
                             const float* __restrict__ gamma, const float* __restrict__ w_init,
                             const float* __restrict__ sigma_init, const float* __restrict__ mu_off,
                             int ei, const float* __restrict__ gates, const float* __restrict__ taus,
                             const float* __restrict__ xi, float* __restrict__ psi) {
    size_t i = (size_t)blockIdx.x * 256 + threadIdx.x;
    if (i >= (size_t)NTOK * 7) return;
    int qc = (int)(i % 7);
    size_t t = i / 7;
    float pe0 = sc[t * 21 + qc * 3 + 0] + mb[t * 21 + qc * 3 + 0] * gamma[qc * 3 + 0];
    float pe1 = sc[t * 21 + qc * 3 + 1] + mb[t * 21 + qc * 3 + 1] * gamma[qc * 3 + 1];
    float pe2 = sc[t * 21 + qc * 3 + 2] + mb[t * 21 + qc * 3 + 2] * gamma[qc * 3 + 2];
    float w  = w_init[ei] + pe0;
    float mu = mu_off[ei] + pe1;
    float z  = sigma_init[ei] + pe2 + taus[t * 35 + ei * 7 + qc];
    float sp = (z > 20.f) ? z : log1pf(__expf(z));
    float sg = sp + 1e-6f;
    float g  = gates[t * 35 + ei * 7 + qc];
    float u  = (xi[i] - mu) / sg;
    psi[i] += g * w * __expf(-0.5f * u * u);
}

// ---------------------------------------------------------------------------
// Host-side orchestration
// ---------------------------------------------------------------------------
#define FPIN(i) ((const float*)d_in[(i)])

struct Blk {
    const float *Wq, *Wk, *Wv, *rescale, *projw, *projb, *pos1, *pos2;
    const float *c1w, *c1b, *c2w, *c2b, *dww, *dwb, *lng, *lnb, *w1, *dw, *w2;
};
struct MsabBufs {
    float *xbuf, *maskb, *bq, *bk, *bv, *bm1, *bt, *bt2, *h128, *h128b;
    float *qnorm, *knorm, *attnb, *wpad;
};

static Blk mkblk(void* const* d_in, int base) {
    Blk p;
    p.Wq = FPIN(base + 0);  p.Wk = FPIN(base + 1);  p.Wv = FPIN(base + 2);
    p.rescale = FPIN(base + 3);
    p.projw = FPIN(base + 4); p.projb = FPIN(base + 5);
    p.pos1 = FPIN(base + 6);  p.pos2 = FPIN(base + 7);
    p.c1w = FPIN(base + 8);   p.c1b = FPIN(base + 9);
    p.c2w = FPIN(base + 10);  p.c2b = FPIN(base + 11);
    p.dww = FPIN(base + 12);  p.dwb = FPIN(base + 13);
    p.lng = FPIN(base + 14);  p.lnb = FPIN(base + 15);
    p.w1 = FPIN(base + 16);   p.dw = FPIN(base + 17);  p.w2 = FPIN(base + 18);
    return p;
}

// pad weights into wpad, then run the WMMA GEMM (compile-time specialized)
static inline void launch_gemm(hipStream_t s, const float* A, int lda, const float* W,
                               const float* b, float* Y, int K, int N, int act,
                               float* wpad) {
    int Kp = (K + 31) & ~31;
    int Np = (N + 15) & ~15;
    int wtot = Np * Kp;
    k_padw<<<dim3((unsigned)((wtot + 255) / 256)), dim3(256), 0, s>>>(W, wpad, N, K, Kp, wtot);
    dim3 g(NTOK / 128), blk(256);
    if (Kp == 32) {
        if (act == 1)      k_gemm_t<32, 1, false><<<g, blk, 0, s>>>(A, lda, wpad, b, Y, N);
        else if (b)        k_gemm_t<32, 0, true ><<<g, blk, 0, s>>>(A, lda, wpad, b, Y, N);
        else               k_gemm_t<32, 0, false><<<g, blk, 0, s>>>(A, lda, wpad, b, Y, N);
    } else if (Kp == 64) {
        k_gemm_t<64, 0, true><<<g, blk, 0, s>>>(A, lda, wpad, b, Y, N);
    } else { // Kp == 128
        k_gemm_t<128, 0, false><<<g, blk, 0, s>>>(A, lda, wpad, b, Y, N);
    }
}

template<int HEADS>
static void run_msab(hipStream_t s, const Blk* blks, int nblk, const MsabBufs& B) {
    const unsigned GR32 = (NTOK * 32 + 255) / 256;
    for (int i = 0; i < nblk; ++i) {
        const Blk& p = blks[i];
        // --- MS-MSA ---
        launch_gemm(s, B.xbuf, 32, p.Wq, nullptr, B.bq, 32, 32, 0, B.wpad);
        launch_gemm(s, B.xbuf, 32, p.Wk, nullptr, B.bk, 32, 32, 0, B.wpad);
        launch_gemm(s, B.xbuf, 32, p.Wv, nullptr, B.bv, 32, 32, 0, B.wpad);
        launch_gemm(s, B.maskb, 32, p.c1w, p.c1b, B.bm1, 32, 32, 0, B.wpad);   // m1
        launch_gemm(s, B.bm1, 32, p.c2w, p.c2b, B.bt, 32, 32, 0, B.wpad);
        k_dwconv_t<32, 5, 2, true><<<GR32, 256, 0, s>>>(B.bt, p.dww, p.dwb, B.bt2); // am
        k_vmul_ma<<<GR32, 256, 0, s>>>(B.bv, B.bm1, B.bt2, (size_t)NTOK * 32);
        k_colnorm<<<128, 256, 0, s>>>(B.bq, B.qnorm);
        k_colnorm<<<128, 256, 0, s>>>(B.bk, B.knorm);
        k_gram<HEADS><<<BATCH * HEADS, 256, 0, s>>>(B.bq, B.bk, B.qnorm, B.knorm,
                                                    p.rescale, B.attnb);
        k_apply_attn<HEADS><<<GR32, 256, 0, s>>>(B.attnb, B.bv, B.bt);
        launch_gemm(s, B.bt, 32, p.projw, p.projb, B.bm1, 32, 32, 0, B.wpad);  // out_c
        k_dwconv_t<32, 3, 1, false><<<GR32, 256, 0, s>>>(B.xbuf, p.pos1, nullptr, B.bt2);
        k_dwconv_t<32, 3, 0, false><<<GR32, 256, 0, s>>>(B.bt2, p.pos2, nullptr, B.bq); // pe
        k_add2<<<GR32, 256, 0, s>>>(B.xbuf, B.bm1, B.bq, (size_t)NTOK * 32);
        // --- FFN ---
        k_layernorm<<<NTOK / 8, 256, 0, s>>>(B.xbuf, p.lng, p.lnb, B.bt);
        launch_gemm(s, B.bt, 32, p.w1, nullptr, B.h128, 32, 128, 1, B.wpad);   // gelu
        k_dwconv_t<128, 3, 1, false><<<(NTOK * 128 + 255) / 256, 256, 0, s>>>(
            B.h128, p.dw, nullptr, B.h128b);
        launch_gemm(s, B.h128b, 128, p.w2, nullptr, B.bt2, 128, 32, 0, B.wpad);
        k_add1<<<GR32, 256, 0, s>>>(B.xbuf, B.bt2, (size_t)NTOK * 32);
    }
}

extern "C" void kernel_launch(void* const* d_in, const int* in_sizes, int n_in,
                              void* d_out, int out_size, void* d_ws, size_t ws_size,
                              hipStream_t stream) {
    (void)in_sizes; (void)n_in; (void)out_size; (void)ws_size;
    const float* x_nchw    = FPIN(0);
    const float* feat_nchw = FPIN(1);

    // ---- carve workspace (all channels-last [token, C]) ----
    float* w = (float*)d_ws;
    size_t off = 0;
    auto carve = [&](int ch) { float* p = w + off; off += (size_t)NTOK * ch; return p; };
    float* xcl    = carve(3);
    float* featcl = carve(32);
    float* xi     = carve(7);
    float* gates  = carve(35);
    float* taus   = carve(35);
    float* psi    = carve(7);
    MsabBufs B;
    B.xbuf = carve(32);  B.maskb = carve(32);
    B.bq   = carve(32);  B.bk    = carve(32);
    B.bv   = carve(32);  B.bm1   = carve(32);
    B.bt   = carve(32);  B.bt2   = carve(32);
    B.h128 = carve(128); B.h128b = carve(128);
    B.qnorm = w + off; off += 128;
    B.knorm = w + off; off += 128;
    B.attnb = w + off; off += 4096;
    B.wpad  = w + off; off += 128 * 128;   // padded-weight scratch (max 64 KB)

    auto gr = [](size_t n) { return dim3((unsigned)((n + 255) / 256)); };

    // ---- layout conversion NCHW -> NHWC ----
    k_tocl<3> <<<gr((size_t)NTOK * 3),  256, 0, stream>>>(x_nchw, xcl);
    k_tocl<32><<<gr((size_t)NTOK * 32), 256, 0, stream>>>(feat_nchw, featcl);

    // ---- basis_attn: xi[tok,7] ----
    launch_gemm(stream, featcl, 32, FPIN(3), FPIN(4), B.bq, 32, 16, 0, B.wpad);
    k_basis<<<gr(NTOK), 256, 0, stream>>>(B.bq, FPIN(2), FPIN(5), xi);

    // ---- orchestrator ----
    k_concat2<64><<<gr((size_t)NTOK * 64), 256, 0, stream>>>(xcl, 3, featcl, 32, B.h128);
    launch_gemm(stream, B.h128, 64, FPIN(6), FPIN(7), B.xbuf, 35, 32, 0, B.wpad);
    k_copy<<<gr((size_t)NTOK * 32), 256, 0, stream>>>(B.maskb, B.xbuf, (size_t)NTOK * 32);
    Blk ob[2] = { mkblk(d_in, 8), mkblk(d_in, 27) };
    run_msab<2>(stream, ob, 2, B);
    launch_gemm(stream, B.xbuf, 32, FPIN(46), FPIN(47), B.h128, 32, 70, 0, B.wpad);
    k_gates<<<gr((size_t)NTOK * 7), 256, 0, stream>>>(B.h128, gates, taus);

    // ---- experts: psi accumulation ----
    k_zero<<<gr((size_t)NTOK * 7), 256, 0, stream>>>(psi, (size_t)NTOK * 7);
    k_concat2<64><<<gr((size_t)NTOK * 64), 256, 0, stream>>>(xi, 7, featcl, 32, B.h128);
    for (int e = 0; e < 5; ++e) {
        int eb = 48 + 9 * e;
        launch_gemm(stream, B.h128, 64, FPIN(eb + 0), FPIN(eb + 1), B.bq, 39, 32, 0, B.wpad);
        k_dwconv_t<32, 3, 3, true><<<gr((size_t)NTOK * 32), 256, 0, stream>>>(
            B.bq, FPIN(eb + 2), FPIN(eb + 3), B.bk);                              // silu
        launch_gemm(stream, B.bk, 32, FPIN(eb + 4), FPIN(eb + 5), B.bv, 32, 21, 0, B.wpad);   // mb
        launch_gemm(stream, B.h128, 64, FPIN(eb + 6), FPIN(eb + 7), B.bm1, 39, 21, 0, B.wpad); // sc
        k_expert_acc<<<gr((size_t)NTOK * 7), 256, 0, stream>>>(
            B.bm1, B.bv, FPIN(eb + 8), FPIN(94), FPIN(95), FPIN(93), e,
            gates, taus, xi, psi);
    }

    // ---- chinet ----
    k_concat3<64><<<gr((size_t)NTOK * 64), 256, 0, stream>>>(xcl, 3, psi, 7, featcl, 32, B.h128);
    launch_gemm(stream, B.h128, 64, FPIN(96), FPIN(97), B.xbuf, 42, 32, 0, B.wpad);
    k_copy<<<gr((size_t)NTOK * 32), 256, 0, stream>>>(B.maskb, B.xbuf, (size_t)NTOK * 32);
    Blk cb[3] = { mkblk(d_in, 98), mkblk(d_in, 117), mkblk(d_in, 136) };
    run_msab<4>(stream, cb, 3, B);
    k_concat2<64><<<gr((size_t)NTOK * 64), 256, 0, stream>>>(xcl, 3, B.xbuf, 32, B.h128);
    launch_gemm(stream, B.h128, 64, FPIN(155), FPIN(156), (float*)d_out, 35, 1, 0, B.wpad);
}
